// NGPRadianceField_16716012716751
// MI455X (gfx1250) — matrix operand order repaired and checked by reading, not compiled, not run on hardware
//
#include <hip/hip_runtime.h>
#include <hip/hip_bf16.h>

typedef __attribute__((ext_vector_type(16))) _Float16 v16h;
typedef __attribute__((ext_vector_type(8)))  _Float16 v8h;
typedef __attribute__((ext_vector_type(8)))  float    v8f;

#define NUM_LEVELS   16
#define LOG2_T       19
#define TABLE_MASK   ((1u << LOG2_T) - 1u)
#define SPB          128     // samples per block
#define THREADS      128     // 4 waves (wave32)
#define ESTR         72      // padded LDS row stride in halves (144 B, 16B-aligned chunks)
#define WSTR         72      // padded LDS column stride for transposed weights

// SCALINGS = floor(16 * (4096/16)^(l/15))
__device__ __constant__ float c_scale[NUM_LEVELS] = {
    16.f, 23.f, 33.f, 48.f, 70.f, 101.f, 147.f, 212.f,
    307.f, 445.f, 645.f, 933.f, 1351.f, 1955.f, 2830.f, 4096.f};

__global__ __launch_bounds__(THREADS)
void ngp_fused_kernel(const float* __restrict__ x,
                      const float* __restrict__ table,   // [16*2^19][2] f32
                      const float* __restrict__ w1,      // [64][64]
                      const float* __restrict__ w2,      // [64][16]
                      float* __restrict__ out,           // [N][16]
                      int n)
{
    __shared__ _Float16 enc [SPB * ESTR];   // f16 encodings, row-major per sample
    __shared__ _Float16 hbuf[SPB * ESTR];   // f16 hidden activations
    __shared__ _Float16 w1T [64 * WSTR];    // w1 transposed: [n][k], K-contiguous
    __shared__ _Float16 w2T [16 * WSTR];    // w2 transposed: [n][k]

    const int tid       = threadIdx.x;
    const int blockBase = blockIdx.x * SPB;
    const int gid       = blockBase + tid;

    // ---- stage transposed f16 weights into LDS ----
    for (int e = tid; e < 64 * 64; e += THREADS) {
        int k = e >> 6, nn = e & 63;                 // w1[k][n]
        w1T[nn * WSTR + k] = (_Float16)w1[e];
    }
    for (int e = tid; e < 64 * 16; e += THREADS) {
        int k = e >> 4, nn = e & 15;                 // w2[k][n]
        w2T[nn * WSTR + k] = (_Float16)w2[e];
    }

    // ---- multires hash encoding: one sample per thread ----
    if (gid < n) {
        const float* xp = x + (size_t)gid * 6;
        float p[6];
        #pragma unroll
        for (int j = 0; j < 6; ++j) p[j] = xp[j];

        #pragma unroll
        for (int pt = 0; pt < 2; ++pt) {
            const float px = p[pt * 3 + 0];
            const float py = p[pt * 3 + 1];
            const float pz = p[pt * 3 + 2];
            #pragma unroll 1   // keep level loop rolled: 8 gathers/level is enough MLP
            for (int l = 0; l < NUM_LEVELS; ++l) {
                const float s  = c_scale[l];
                const float vx = px * s, vy = py * s, vz = pz * s;
                const float fxf = floorf(vx), fyf = floorf(vy), fzf = floorf(vz);
                const unsigned fx = (unsigned)(int)fxf;
                const unsigned fy = (unsigned)(int)fyf;
                const unsigned fz = (unsigned)(int)fzf;
                const unsigned cx = (unsigned)(int)ceilf(vx);
                const unsigned cy = (unsigned)(int)ceilf(vy);
                const unsigned cz = (unsigned)(int)ceilf(vz);
                const float wx = vx - fxf, wy = vy - fyf, wz = vz - fzf;
                const unsigned lvl_off = (unsigned)l << LOG2_T;

                // CORNER_MASK rows encoded as bit0=x, bit1=y, bit2=z (1 => ceil)
                const unsigned cm[8] = {7u, 5u, 4u, 6u, 3u, 1u, 0u, 2u};
                float fc[8][2];
                #pragma unroll
                for (int c = 0; c < 8; ++c) {
                    const unsigned X = (cm[c] & 1u) ? cx : fx;
                    const unsigned Y = (cm[c] & 2u) ? cy : fy;
                    const unsigned Z = (cm[c] & 4u) ? cz : fz;
                    unsigned hsh = (X ^ (Y * 2654435761u) ^ (Z * 805459861u)) & TABLE_MASK;
                    const float* tp = table + (size_t)(hsh + lvl_off) * 2u;
                    fc[c][0] = tp[0];
                    fc[c][1] = tp[1];
                }
                #pragma unroll
                for (int f = 0; f < 2; ++f) {
                    const float f03   = fc[0][f] * wx + fc[3][f] * (1.f - wx);
                    const float f12   = fc[1][f] * wx + fc[2][f] * (1.f - wx);
                    const float f56   = fc[5][f] * wx + fc[6][f] * (1.f - wx);
                    const float f47   = fc[4][f] * wx + fc[7][f] * (1.f - wx);
                    const float f0312 = f03 * wy + f12 * (1.f - wy);
                    const float f4756 = f47 * wy + f56 * (1.f - wy);
                    const float ev    = f0312 * wz + f4756 * (1.f - wz);
                    enc[tid * ESTR + pt * 32 + l * 2 + f] = (_Float16)ev;
                }
            }
        }
    } else {
        #pragma unroll
        for (int j = 0; j < 64; ++j) enc[tid * ESTR + j] = (_Float16)0.f;
    }
    __syncthreads();

    // ---- fused MLP via WMMA: each wave owns 32 sample rows (2 M-tiles) ----
    const int wave = tid >> 5;
    const int lane = tid & 31;
    const int hs   = lane >> 4;      // half-wave select (K split per ISA layout)
    const int lc   = lane & 15;      // row (A) / column (B,D) index

    #pragma unroll
    for (int mt = 0; mt < 2; ++mt) {
        const int mbase = wave * 32 + mt * 16;   // first sample row of this tile
        const int arow  = mbase + lc;

        // layer 1: h = relu(enc @ w1), N=64 -> 4 N-tiles, K=64 -> 2 k-steps
        #pragma unroll
        for (int nt = 0; nt < 4; ++nt) {
            v8f acc = {0.f, 0.f, 0.f, 0.f, 0.f, 0.f, 0.f, 0.f};
            #pragma unroll
            for (int ks = 0; ks < 2; ++ks) {
                // A: lanes<16 take K {0..7,16..23}; lanes>=16 take K {8..15,24..31}
                const _Float16* ap = &enc[arow * ESTR + ks * 32 + hs * 8];
                v8h a0 = *(const v8h*)ap;
                v8h a1 = *(const v8h*)(ap + 16);
                v16h A;
                #pragma unroll
                for (int j = 0; j < 8; ++j) { A[j] = a0[j]; A[j + 8] = a1[j]; }
                // B: lanes<16 hold K {0..7,16..23}, lanes>=16 hold K {8..15,24..31} of column lc
                const _Float16* bp = &w1T[(nt * 16 + lc) * WSTR + ks * 32 + hs * 8];
                v8h b0 = *(const v8h*)bp;
                v8h b1 = *(const v8h*)(bp + 16);
                v16h B;
                #pragma unroll
                for (int j = 0; j < 8; ++j) { B[j] = b0[j]; B[j + 8] = b1[j]; }
                acc = __builtin_amdgcn_wmma_f32_16x16x32_f16(
                    false, A, false, B, (short)0, acc, false, false);
            }
            // D layout: VGPR r -> row r (lanes<16) / row r+8 (lanes>=16), col lc
            #pragma unroll
            for (int r = 0; r < 8; ++r) {
                const float v = fmaxf(acc[r], 0.f);
                hbuf[(mbase + r + hs * 8) * ESTR + nt * 16 + lc] = (_Float16)v;
            }
        }

        // layer 2: out = h @ w2, N=16, K=64 -> 2 k-steps
        v8f acc2 = {0.f, 0.f, 0.f, 0.f, 0.f, 0.f, 0.f, 0.f};
        #pragma unroll
        for (int ks = 0; ks < 2; ++ks) {
            const _Float16* ap = &hbuf[arow * ESTR + ks * 32 + hs * 8];
            v8h a0 = *(const v8h*)ap;
            v8h a1 = *(const v8h*)(ap + 16);
            v16h A;
            #pragma unroll
            for (int j = 0; j < 8; ++j) { A[j] = a0[j]; A[j + 8] = a1[j]; }
            const _Float16* bp = &w2T[lc * WSTR + ks * 32 + hs * 8];
            v8h b0 = *(const v8h*)bp;
            v8h b1 = *(const v8h*)(bp + 16);
            v16h B;
            #pragma unroll
            for (int j = 0; j < 8; ++j) { B[j] = b0[j]; B[j + 8] = b1[j]; }
            acc2 = __builtin_amdgcn_wmma_f32_16x16x32_f16(
                false, A, false, B, (short)0, acc2, false, false);
        }

        // epilogue: col 0 -> exp(v - 1); store 16 f32 per sample row
        #pragma unroll
        for (int r = 0; r < 8; ++r) {
            const int row = blockBase + mbase + r + hs * 8;
            if (row < n) {
                float v = acc2[r];
                if (lc == 0) v = expf(v - 1.f);
                out[(size_t)row * 16 + lc] = v;
            }
        }
    }
}

extern "C" void kernel_launch(void* const* d_in, const int* in_sizes, int n_in,
                              void* d_out, int out_size, void* d_ws, size_t ws_size,
                              hipStream_t stream) {
    const float* x     = (const float*)d_in[0];
    const float* table = (const float*)d_in[1];
    const float* w1    = (const float*)d_in[2];
    const float* w2    = (const float*)d_in[3];
    float* out = (float*)d_out;
    const int n = in_sizes[0] / 6;
    const int blocks = (n + SPB - 1) / SPB;
    ngp_fused_kernel<<<blocks, THREADS, 0, stream>>>(x, table, w1, w2, out, n);
}
